// MoETransformer_14972255994102
// MI455X (gfx1250) — compile-verified, matrix operand order checked by reading, not run on hardware
//
#include <hip/hip_runtime.h>
#include <math.h>

// ---------------------------------------------------------------------------
// MoE encoder-decoder transformer forward for MI455X (gfx1250, wave32).
// Heavy math on v_wmma_f32_16x16x32_f16 with f32 accumulation.
// Tile staging uses CDNA5 async global->LDS copies (ASYNCcnt) with
// double-buffered LDS so the next K-step's tiles land while WMMAs run.
// ---------------------------------------------------------------------------

typedef _Float16 half_t;
typedef __attribute__((ext_vector_type(16))) _Float16 v16h;
typedef __attribute__((ext_vector_type(8)))  float    v8f;

constexpr int kV  = 32000;
constexpr int kD  = 512;
constexpr int kH  = 8;
constexpr int kHD = kD / kH;     // 64
constexpr int kF  = 2048;
constexpr int kE  = 8;
constexpr int kL  = 6;
constexpr int kB  = 4;
constexpr int kS  = 512;
constexpr int kT  = 512;
constexpr int kM  = kB * kS;     // 2048 tokens (B*S == B*T)

static_assert(kM % 128 == 0 && kD % 64 == 0 && kV % 64 == 0, "tile divisibility");

// ---------------------------- WMMA helpers --------------------------------

__device__ __forceinline__ v8f wmma_f32(v16h a, v16h b, v8f c) {
  // D(f32 16x16) = A(f16 16x32) * B(f16 32x16) + C
  return __builtin_amdgcn_wmma_f32_16x16x32_f16(false, a, false, b, (short)0, c,
                                                false, false);
}

// A/B operand fragment from a K-major (row-contiguous) row.
// rowbase points at (row, k0). Lane-dependent koff = (lane>=16)?8:0 selects the
// documented wave32 striping: halves [0..7] = K koff..koff+7, [8..15] = +16.
__device__ __forceinline__ v16h frag_h(const half_t* rowbase, int koff) {
  v16h f;
#pragma unroll
  for (int i = 0; i < 8; ++i) {
    f[i]     = rowbase[koff + i];
    f[i + 8] = rowbase[koff + 16 + i];
  }
  return f;
}

__device__ __forceinline__ v16h frag_f32(const float* rowbase, int koff) {
  v16h f;
#pragma unroll
  for (int i = 0; i < 8; ++i) {
    f[i]     = (half_t)rowbase[koff + i];
    f[i + 8] = (half_t)rowbase[koff + 16 + i];
  }
  return f;
}

__device__ __forceinline__ v8f zero8() {
  v8f c;
#pragma unroll
  for (int i = 0; i < 8; ++i) c[i] = 0.0f;
  return c;
}

// -------------------- CDNA5 async global -> LDS copy ------------------------
// Per-lane 16-byte copy from global memory into LDS, tracked by ASYNCcnt.
// LDS offset = low 32 bits of the generic pointer (flat-shared aperture maps
// LDS at addr[31:0] per the ISA aperture rules).
__device__ __forceinline__ void async_copy16(void* lds_dst, const void* gsrc) {
  const unsigned loff = (unsigned)(uintptr_t)lds_dst;
  asm volatile("global_load_async_to_lds_b128 %0, %1, off"
               :
               : "v"(loff), "v"(gsrc)
               : "memory");
}
__device__ __forceinline__ void async_wait0() {
  asm volatile("s_wait_asynccnt 0x0" ::: "memory");
}

// ------------------------------- GEMM --------------------------------------
// Out[m,n] = sum_k A_h[m,k] * W_f32[n,k] + bias[n]
// MODE 0: store fp32 [M,N];  MODE 1: store f16 [M,N];
// MODE 2: store f16 transposed as vT[b][n][s] with m = b*kS + s (for V).
// Double-buffered LDS: A tile staged via async-to-LDS, W tile (fp32->f16)
// staged via registers; K-step k+1 is in flight while k's WMMAs execute.
template<int MODE>
__global__ void gemm_kernel(const half_t* __restrict__ A, const float* __restrict__ W,
                            const float* __restrict__ bias, void* __restrict__ OutP,
                            int M, int N, int Kd) {
  constexpr int BM = 128, BN = 64, BK = 32, PIT = BK + 8;
  __shared__ half_t As[2][BM][PIT];
  __shared__ half_t Ws[2][BN][PIT];

  const int tid  = threadIdx.x, wid = tid >> 5, lane = tid & 31;
  const int m0   = blockIdx.x * BM, n0 = blockIdx.y * BN;
  const int r    = lane & 15, koff = (lane >> 4) << 3;

  v8f acc[4];
#pragma unroll
  for (int nt = 0; nt < 4; ++nt) acc[nt] = zero8();

  // A tile: 128x32 halves = 512 x 16B chunks; thread copies rows arow, arow+64.
  const int arow = tid >> 2, acol = (tid & 3) << 3;
  // W tile: 64x32 fp32; thread owns 8 consecutive floats.
  const int wrow = tid >> 2, wcol = (tid & 3) << 3;

  const int NK = Kd / BK;

  // ---- prologue: stage K-step 0 into buffer 0 ----
  async_copy16(&As[0][arow][acol],      A + (size_t)(m0 + arow) * Kd + acol);
  async_copy16(&As[0][arow + 64][acol], A + (size_t)(m0 + arow + 64) * Kd + acol);
  {
    const float* ws = W + (size_t)(n0 + wrow) * Kd + wcol;
#pragma unroll
    for (int i = 0; i < 8; ++i) Ws[0][wrow][wcol + i] = (half_t)ws[i];
  }
  async_wait0();
  __syncthreads();

  for (int kk = 0; kk < NK; ++kk) {
    const int cur = kk & 1, nxt = cur ^ 1;
    const bool more = (kk + 1 < NK);
    float wr[8];
    if (more) {  // launch next tile's traffic before this tile's math
      const int kn = (kk + 1) * BK;
      async_copy16(&As[nxt][arow][acol],      A + (size_t)(m0 + arow) * Kd + kn + acol);
      async_copy16(&As[nxt][arow + 64][acol], A + (size_t)(m0 + arow + 64) * Kd + kn + acol);
      const float* ws = W + (size_t)(n0 + wrow) * Kd + kn + wcol;
      __builtin_prefetch(ws + BK, 0, 1);  // K-step k+2 weight row
#pragma unroll
      for (int i = 0; i < 8; ++i) wr[i] = ws[i];
    }

    v16h a = frag_h(&As[cur][wid * 16 + r][0], koff);
#pragma unroll
    for (int nt = 0; nt < 4; ++nt) {
      v16h b = frag_h(&Ws[cur][nt * 16 + r][0], koff);
      acc[nt] = wmma_f32(a, b, acc[nt]);
    }

    if (more) {
#pragma unroll
      for (int i = 0; i < 8; ++i) Ws[nxt][wrow][wcol + i] = (half_t)wr[i];
    }
    async_wait0();
    __syncthreads();
  }

  const int mbase = m0 + wid * 16 + ((lane >> 4) << 3);
#pragma unroll
  for (int nt = 0; nt < 4; ++nt) {
    const int n = n0 + nt * 16 + r;
    const float bv = bias[n];
#pragma unroll
    for (int j = 0; j < 8; ++j) {
      const int m = mbase + j;
      const float v = acc[nt][j] + bv;
      if (MODE == 0) {
        ((float*)OutP)[(size_t)m * N + n] = v;
      } else if (MODE == 1) {
        ((half_t*)OutP)[(size_t)m * N + n] = (half_t)v;
      } else {
        const int bb = m >> 9, ss = m & (kS - 1);   // m = bb*kS + ss
        ((half_t*)OutP)[((size_t)bb * kD + n) * kS + ss] = (half_t)v;
      }
    }
  }
}

// ----------------------------- Attention -----------------------------------
constexpr int SCPIT = kS + 8;    // fp32 score row pitch
constexpr int PRPIT = kS + 16;   // f16 prob row pitch
constexpr size_t ATTN_SHM = (size_t)64 * SCPIT * 4 + (size_t)64 * PRPIT * 2;

__global__ void attn_kernel(const half_t* __restrict__ qh, const half_t* __restrict__ kh,
                            const half_t* __restrict__ vT, half_t* __restrict__ ctx,
                            int causal) {
  extern __shared__ char smem[];
  float*  sc = (float*)smem;                              // [64][SCPIT]
  half_t* pr = (half_t*)(smem + (size_t)64 * SCPIT * 4);  // [64][PRPIT]

  const int tid  = threadIdx.x, wid = tid >> 5, lane = tid & 31;
  const int qblk = blockIdx.x & 7;            // 8 q-blocks of 64
  const int h    = (blockIdx.x >> 3) & 7;     // 8 heads
  const int bb   = blockIdx.x >> 6;           // batch
  const int q0   = qblk * 64 + wid * 16;      // this wave's 16 query rows
  const int r    = lane & 15, koff = (lane >> 4) << 3;

  const half_t* qbase = qh + (size_t)(bb * kT + q0 + r) * kD + h * kHD;
  v16h qa0 = frag_h(qbase, koff);         // K = 0..31 of head dim
  v16h qa1 = frag_h(qbase + 32, koff);    // K = 32..63

  // scores = Q K^T / sqrt(HD)  (+ causal mask)
  for (int kt = 0; kt < kS / 16; ++kt) {
    const half_t* kbase = kh + (size_t)(bb * kS + kt * 16 + r) * kD + h * kHD;
    v16h kb0 = frag_h(kbase, koff);
    v16h kb1 = frag_h(kbase + 32, koff);
    v8f c = zero8();
    c = wmma_f32(qa0, kb0, c);
    c = wmma_f32(qa1, kb1, c);
    const int n = kt * 16 + r;  // key index
#pragma unroll
    for (int j = 0; j < 8; ++j) {
      const int m = ((lane >> 4) << 3) + j;  // local query row
      float s = c[j] * 0.125f;               // 1/sqrt(64)
      if (causal && n > q0 + m) s = -INFINITY;
      sc[(wid * 16 + m) * SCPIT + n] = s;
    }
  }
  __syncthreads();

  {  // row softmax: 2 lanes per query row
    const int row = lane >> 1, hf = lane & 1;
    float* srow = &sc[(wid * 16 + row) * SCPIT + hf * (kS / 2)];
    float mx = -INFINITY;
    for (int j = 0; j < kS / 2; ++j) mx = fmaxf(mx, srow[j]);
    mx = fmaxf(mx, __shfl_xor(mx, 1, 32));
    float sum = 0.f;
    for (int j = 0; j < kS / 2; ++j) { float e = __expf(srow[j] - mx); srow[j] = e; sum += e; }
    sum += __shfl_xor(sum, 1, 32);
    const float inv = 1.f / sum;
    half_t* prow = &pr[(wid * 16 + row) * PRPIT + hf * (kS / 2)];
    for (int j = 0; j < kS / 2; ++j) prow[j] = (half_t)(srow[j] * inv);
  }
  __syncthreads();

  // out = P V   (B operand from transposed V: vT[b][d][s], K = s contiguous)
#pragma unroll
  for (int nt = 0; nt < 4; ++nt) {
    v8f c = zero8();
    for (int st = 0; st < kS / 32; ++st) {
      v16h a   = frag_h(&pr[(wid * 16 + r) * PRPIT + st * 32], koff);
      v16h bfr = frag_h(vT + ((size_t)bb * kD + h * kHD + nt * 16 + r) * kS + st * 32, koff);
      c = wmma_f32(a, bfr, c);
    }
#pragma unroll
    for (int j = 0; j < 8; ++j) {
      const int m = ((lane >> 4) << 3) + j;
      ctx[(size_t)(bb * kT + q0 + m) * kD + h * kHD + nt * 16 + r] = (half_t)c[j];
    }
  }
}

// ------------------------------- MoE FFN -----------------------------------
// One 16-token tile per block; loops all E experts densely (matches reference).
// h = relu(x W1^T + b1) kept in LDS;  y += combine[t,e] * (h W2^T + b2).
constexpr int XP = kD + 8;
constexpr int HP = kF + 8;
constexpr size_t MOE_SHM = (size_t)16 * XP * 2 + (size_t)16 * HP * 2 +
                           (size_t)16 * kD * 4 + (size_t)16 * kE * 4;

__global__ void moe_kernel(const half_t* __restrict__ xh,
                           const float* __restrict__ w1, const float* __restrict__ b1,
                           const float* __restrict__ w2, const float* __restrict__ b2,
                           const float* __restrict__ comb, float* __restrict__ outp) {
  extern __shared__ char smem[];
  half_t* xs  = (half_t*)smem;                // [16][XP]  f16 x tile
  half_t* hs  = xs + 16 * XP;                 // [16][HP]  f16 hidden tile
  float*  ys  = (float*)(hs + 16 * HP);       // [16][kD]  fp32 accum
  float*  cbm = ys + 16 * kD;                 // [16][kE]

  const int tid = threadIdx.x, wid = tid >> 5, lane = tid & 31;
  const int t0  = blockIdx.x * 16;
  const int r   = lane & 15, koff = (lane >> 4) << 3;

  // async-stage the 16x512 f16 x tile (1024 x 16B chunks, 4 per thread)
  for (int c = tid; c < 1024; c += 256) {
    const int row = c >> 6, col = (c & 63) << 3;
    async_copy16(&xs[row * XP + col], xh + (size_t)(t0 + row) * kD + col);
  }
  for (int i = tid; i < 16 * kD; i += 256) ys[i] = 0.f;
  for (int i = tid; i < 16 * kE; i += 256) cbm[i] = comb[(size_t)t0 * kE + i];
  async_wait0();
  __syncthreads();

  for (int e = 0; e < kE; ++e) {
    const float* W1 = w1 + (size_t)e * kF * kD;
    const float* B1 = b1 + (size_t)e * kF;
    for (int nt = wid; nt < kF / 16; nt += 8) {          // 16 n-tiles per wave
      v8f c = zero8();
      const float* wrow = W1 + (size_t)(nt * 16 + r) * kD;
      for (int k0 = 0; k0 < kD; k0 += 32) {
        v16h a   = frag_h(&xs[r * XP + k0], koff);
        v16h bfr = frag_f32(wrow + k0, koff);
        c = wmma_f32(a, bfr, c);
      }
      const int n = nt * 16 + r;
      const float bv = B1[n];
#pragma unroll
      for (int j = 0; j < 8; ++j) {
        const int m = ((lane >> 4) << 3) + j;
        hs[m * HP + n] = (half_t)fmaxf(c[j] + bv, 0.f);   // ReLU
      }
    }
    __syncthreads();

    const float* W2 = w2 + (size_t)e * kD * kF;
    const float* B2 = b2 + (size_t)e * kD;
    for (int nt = wid; nt < kD / 16; nt += 8) {          // 4 n-tiles per wave
      v8f c = zero8();
      const float* wrow = W2 + (size_t)(nt * 16 + r) * kF;
      for (int k0 = 0; k0 < kF; k0 += 32) {
        v16h a   = frag_h(&hs[r * HP + k0], koff);
        v16h bfr = frag_f32(wrow + k0, koff);
        c = wmma_f32(a, bfr, c);
      }
      const int n = nt * 16 + r;
      const float bv = B2[n];
#pragma unroll
      for (int j = 0; j < 8; ++j) {
        const int m = ((lane >> 4) << 3) + j;
        ys[m * kD + n] += cbm[m * kE + e] * (c[j] + bv);  // unique (m,n) writer
      }
    }
    __syncthreads();
  }

  for (int i = tid; i < 16 * kD; i += 256)
    outp[(size_t)(t0 + i / kD) * kD + (i % kD)] = ys[i];
}

// ----------------------- small supporting kernels ---------------------------

__global__ void embed_kernel(const int* __restrict__ tok, const float* __restrict__ emb,
                             const float* __restrict__ pe, float* __restrict__ xf,
                             half_t* __restrict__ xh, int seqlen) {
  const int idx  = blockIdx.x;        // flat token index
  const int tpos = idx % seqlen;
  const int tk   = tok[idx];
  for (int d = threadIdx.x; d < kD; d += blockDim.x) {
    const float v = emb[(size_t)tk * kD + d] * 22.62741699796952f /* sqrt(512) */
                  + pe[(size_t)tpos * kD + d];
    xf[(size_t)idx * kD + d] = v;
    xh[(size_t)idx * kD + d] = (half_t)v;
  }
}

__global__ void add_ln_kernel(float* __restrict__ xf, const float* __restrict__ res,
                              const float* __restrict__ gg, const float* __restrict__ bb,
                              half_t* __restrict__ xh) {
  __shared__ float sred[256];
  const int t = blockIdx.x, tid = threadIdx.x;
  const int d0 = tid, d1 = tid + 256;
  float v0 = xf[(size_t)t * kD + d0] + res[(size_t)t * kD + d0];
  float v1 = xf[(size_t)t * kD + d1] + res[(size_t)t * kD + d1];

  sred[tid] = v0 + v1;
  __syncthreads();
  for (int s2 = 128; s2 > 0; s2 >>= 1) { if (tid < s2) sred[tid] += sred[tid + s2]; __syncthreads(); }
  const float mu = sred[0] * (1.f / kD);
  __syncthreads();

  const float a0 = v0 - mu, a1 = v1 - mu;
  sred[tid] = a0 * a0 + a1 * a1;
  __syncthreads();
  for (int s2 = 128; s2 > 0; s2 >>= 1) { if (tid < s2) sred[tid] += sred[tid + s2]; __syncthreads(); }
  const float rstd = rsqrtf(sred[0] * (1.f / kD) + 1e-5f);

  const float o0 = a0 * rstd * gg[d0] + bb[d0];
  const float o1 = a1 * rstd * gg[d1] + bb[d1];
  xf[(size_t)t * kD + d0] = o0;  xf[(size_t)t * kD + d1] = o1;
  xh[(size_t)t * kD + d0] = (half_t)o0;  xh[(size_t)t * kD + d1] = (half_t)o1;
}

// top-2 softmax router; accumulates per-expert mean-prob and dispatch counts.
__global__ void router_kernel(const float* __restrict__ xf, const float* __restrict__ rw,
                              const float* __restrict__ rb, float* __restrict__ comb,
                              float* __restrict__ stats /* [psum x8, cnt x8] */) {
  __shared__ float sred[256];
  __shared__ float logit[kE];
  const int t = blockIdx.x, tid = threadIdx.x;

  float part[kE];
#pragma unroll
  for (int e = 0; e < kE; ++e) part[e] = 0.f;
  for (int d = tid; d < kD; d += 256) {
    const float xv = xf[(size_t)t * kD + d];
#pragma unroll
    for (int e = 0; e < kE; ++e) part[e] += xv * rw[(size_t)e * kD + d];
  }
  for (int e = 0; e < kE; ++e) {
    sred[tid] = part[e];
    __syncthreads();
    for (int s2 = 128; s2 > 0; s2 >>= 1) { if (tid < s2) sred[tid] += sred[tid + s2]; __syncthreads(); }
    if (tid == 0) logit[e] = sred[0] + rb[e];
    __syncthreads();
  }
  if (tid == 0) {
    float mx = logit[0];
    for (int e = 1; e < kE; ++e) mx = fmaxf(mx, logit[e]);
    float p[kE], sum = 0.f;
    for (int e = 0; e < kE; ++e) { p[e] = __expf(logit[e] - mx); sum += p[e]; }
    const float inv = 1.f / sum;
    for (int e = 0; e < kE; ++e) { p[e] *= inv; atomicAdd(&stats[e], p[e]); }
    int i1 = 0;
    for (int e = 1; e < kE; ++e) if (p[e] > p[i1]) i1 = e;         // ties -> lower idx
    int i2 = (i1 == 0) ? 1 : 0;
    for (int e = 0; e < kE; ++e) if (e != i1 && p[e] > p[i2]) i2 = e;
    const float wsum = p[i1] + p[i2];
    for (int e = 0; e < kE; ++e) comb[(size_t)t * kE + e] = 0.f;
    comb[(size_t)t * kE + i1] = p[i1] / wsum;
    comb[(size_t)t * kE + i2] = p[i2] / wsum;
    atomicAdd(&stats[kE + i1], 1.f);
    atomicAdd(&stats[kE + i2], 1.f);
  }
}

__global__ void zero_kernel(float* p, int n) {
  const int i = blockIdx.x * blockDim.x + threadIdx.x;
  if (i < n) p[i] = 0.f;
}

__global__ void lb_kernel(const float* __restrict__ stats, float* __restrict__ out) {
  if (threadIdx.x == 0 && blockIdx.x == 0) {
    float lb = 0.f;
    for (int l = 0; l < 2 * kL; ++l) {
      const float* st = stats + l * 16;
      float acc = 0.f;
      for (int e = 0; e < kE; ++e) {
        const float P = st[e] / (float)kM;                 // mean prob over B*T
        const float f = st[kE + e] / ((float)kM * 2.f);    // mean one_hot over B*T*K
        acc += f * P;
      }
      lb += (float)kE * acc;
    }
    out[0] = lb;
  }
}

// ------------------------------- host --------------------------------------

extern "C" void kernel_launch(void* const* d_in, const int* in_sizes, int n_in,
                              void* d_out, int out_size, void* d_ws, size_t ws_size,
                              hipStream_t stream) {
  (void)in_sizes; (void)n_in; (void)out_size; (void)ws_size;
  auto fin = [&](int i) { return (const float*)d_in[i]; };

  // JAX pytree (sorted-key) flatten order:
  //   0: dec_emb | 1..168: dec_layers (6 x 28) | 169: enc_emb |
  //   170..277: enc_layers (6 x 18) | 278: out_b | 279: out_w | 280: pe |
  //   281: src | 282: tgt
  const float* dec_emb = fin(0);
  const float* enc_emb = fin(169);
  const float* out_b   = fin(278);
  const float* out_w   = fin(279);
  const float* pe      = fin(280);
  const int*   srcTok  = (const int*)d_in[281];
  const int*   tgtTok  = (const int*)d_in[282];

  // workspace carve (~21 MB)
  char* wp = (char*)d_ws;
  auto carve = [&](size_t bytes) { void* p = (void*)wp; wp += (bytes + 255) & ~(size_t)255; return p; };
  float*  xf    = (float*) carve((size_t)kM * kD * 4);
  float*  tmpf  = (float*) carve((size_t)kM * kD * 4);
  half_t* xh    = (half_t*)carve((size_t)kM * kD * 2);
  half_t* memh  = (half_t*)carve((size_t)kM * kD * 2);
  half_t* qh    = (half_t*)carve((size_t)kM * kD * 2);
  half_t* kh    = (half_t*)carve((size_t)kM * kD * 2);
  half_t* vTh   = (half_t*)carve((size_t)kM * kD * 2);
  half_t* ctxh  = (half_t*)carve((size_t)kM * kD * 2);
  float*  comb  = (float*) carve((size_t)kM * kE * 4);
  float*  stats = (float*) carve((size_t)2 * kL * 16 * 4);

  zero_kernel<<<1, 256, 0, stream>>>(stats, 2 * kL * 16);

  const dim3 gproj(kM / 128, kD / 64);
  auto run_mha = [&](const half_t* aq, const half_t* akv,
                     const float* wq, const float* bq, const float* wk, const float* bk,
                     const float* wv, const float* bv, const float* wo, const float* bo,
                     int causal) {
    gemm_kernel<1><<<gproj, 256, 0, stream>>>(aq,  wq, bq, qh,  kM, kD, kD);
    gemm_kernel<1><<<gproj, 256, 0, stream>>>(akv, wk, bk, kh,  kM, kD, kD);
    gemm_kernel<2><<<gproj, 256, 0, stream>>>(akv, wv, bv, vTh, kM, kD, kD);
    attn_kernel<<<dim3(kB * kH * (kT / 64)), 128, ATTN_SHM, stream>>>(qh, kh, vTh, ctxh, causal);
    gemm_kernel<0><<<gproj, 256, 0, stream>>>(ctxh, wo, bo, tmpf, kM, kD, kD);
  };
  auto run_moe_block = [&](const float* rw, const float* rb,
                           const float* w1p, const float* b1p,
                           const float* w2p, const float* b2p,
                           const float* lg, const float* lbb, float* statSlot) {
    router_kernel<<<kM, 256, 0, stream>>>(xf, rw, rb, comb, statSlot);
    moe_kernel<<<kM / 16, 256, MOE_SHM, stream>>>(xh, w1p, b1p, w2p, b2p, comb, tmpf);
    add_ln_kernel<<<kM, 256, 0, stream>>>(xf, tmpf, lg, lbb, xh);
  };

  // -------- encoder --------
  embed_kernel<<<kM, 256, 0, stream>>>(srcTok, enc_emb, pe, xf, xh, kS);
  for (int l = 0; l < kL; ++l) {
    const int base = 170 + 18 * l;
    // sorted: b1=0 b2=1 ln1_b=2 ln1_g=3 ln2_b=4 ln2_g=5 router_b=6 router_w=7
    //         sa_bk=8 sa_bo=9 sa_bq=10 sa_bv=11 sa_wk=12 sa_wo=13 sa_wq=14 sa_wv=15 w1=16 w2=17
    run_mha(xh, xh, fin(base + 14), fin(base + 10), fin(base + 12), fin(base + 8),
            fin(base + 15), fin(base + 11), fin(base + 13), fin(base + 9), /*causal=*/0);
    add_ln_kernel<<<kM, 256, 0, stream>>>(xf, tmpf, fin(base + 3), fin(base + 2), xh);
    run_moe_block(fin(base + 7), fin(base + 6), fin(base + 16), fin(base + 0),
                  fin(base + 17), fin(base + 1), fin(base + 5), fin(base + 4),
                  stats + l * 16);
  }
  hipMemcpyAsync(memh, xh, (size_t)kM * kD * sizeof(half_t),
                 hipMemcpyDeviceToDevice, stream);

  // -------- decoder --------
  embed_kernel<<<kM, 256, 0, stream>>>(tgtTok, dec_emb, pe, xf, xh, kT);
  for (int l = 0; l < kL; ++l) {
    const int base = 1 + 28 * l;
    // sorted: b1=0 b2=1 ca_bk=2 ca_bo=3 ca_bq=4 ca_bv=5 ca_wk=6 ca_wo=7 ca_wq=8 ca_wv=9
    //         ln1_b=10 ln1_g=11 ln2_b=12 ln2_g=13 ln3_b=14 ln3_g=15 router_b=16 router_w=17
    //         sa_bk=18 sa_bo=19 sa_bq=20 sa_bv=21 sa_wk=22 sa_wo=23 sa_wq=24 sa_wv=25 w1=26 w2=27
    run_mha(xh, xh, fin(base + 24), fin(base + 20), fin(base + 22), fin(base + 18),
            fin(base + 25), fin(base + 21), fin(base + 23), fin(base + 19), /*causal=*/1);
    add_ln_kernel<<<kM, 256, 0, stream>>>(xf, tmpf, fin(base + 11), fin(base + 10), xh);
    run_mha(xh, memh, fin(base + 8), fin(base + 4), fin(base + 6), fin(base + 2),
            fin(base + 9), fin(base + 5), fin(base + 7), fin(base + 3), /*causal=*/0);
    add_ln_kernel<<<kM, 256, 0, stream>>>(xf, tmpf, fin(base + 13), fin(base + 12), xh);
    run_moe_block(fin(base + 17), fin(base + 16), fin(base + 26), fin(base + 0),
                  fin(base + 27), fin(base + 1), fin(base + 15), fin(base + 14),
                  stats + (kL + l) * 16);
  }

  // -------- vocab projection + load-balance loss --------
  gemm_kernel<0><<<dim3(kM / 128, kV / 64), 256, 0, stream>>>(
      xh, out_w, out_b, (float*)d_out, kM, kV, kD);
  lb_kernel<<<1, 1, 0, stream>>>(stats, (float*)d_out + (size_t)kM * kV);
}
